// GAT_52046413693564
// MI455X (gfx1250) — matrix operand order repaired and checked by reading, not compiled, not run on hardware
//
#include <hip/hip_runtime.h>
#include <hip/hip_bf16.h>

typedef __attribute__((ext_vector_type(16))) _Float16 v16h;
typedef __attribute__((ext_vector_type(8)))  _Float16 v8h;
typedef __attribute__((ext_vector_type(4)))  _Float16 v4h;
typedef __attribute__((ext_vector_type(8)))  float    v8f;

#define GN     4096   // nodes
#define GNFEAT 2048
#define GD     256    // hidden per head
#define GC     512    // nclass
#define GALPHA 0.2f

// ---------------------------------------------------------------------------
// Generic WMMA GEMM: C[M x NC] = A[M x K] * B[K x NC], with B given transposed
// (BT[NC x K], row-major).  Block = 256 threads = 8 waves; each wave computes a
// 16 x 128 strip (8 accumulator tiles of 16x16).  grid = (M/128, NC/128).
// MODE 0: store f16 transposed  out[col*outStride + row]        (Wh^T)
// MODE 1: store f16 row-major with double-ELU at column offset  (concat layers)
// MODE 2: store f32 row-major                                   (output heads)
// ---------------------------------------------------------------------------
template <int MODE>
__global__ __launch_bounds__(256)
void gemm_wmma(const _Float16* __restrict__ A, const _Float16* __restrict__ BT,
               void* __restrict__ outp, int M, int K, int outStride, int colOffset)
{
    const int lane = threadIdx.x & 31;
    const int wave = threadIdx.x >> 5;
    const int hi   = lane >> 4;     // lane group 0/1
    const int lp   = lane & 15;
    const int rowBase  = blockIdx.x * 128 + wave * 16;
    const int colChunk = blockIdx.y * 128;

    v8f acc[8] = {};

    const _Float16* Arow = A + (size_t)(rowBase + lp) * (size_t)K;
    for (int kb = 0; kb < K; kb += 32) {
        // 16-bit A fragment as this code loads it (16x32, contiguous k order, same as B):
        //   lanes 0-15: halves 0..15 -> K kb+0..15
        //   lanes16-31: halves 0..15 -> K kb+16..31
        v16h af = *(const v16h*)(Arow + kb + hi * 16);
#pragma unroll
        for (int t = 0; t < 8; ++t) {
            // 16-bit B fragment (32x16): lane = column, 16 contiguous K halves
            // starting at kb + (lane group)*16
            const _Float16* Brow =
                BT + (size_t)(colChunk + t * 16 + lp) * (size_t)K + kb + hi * 16;
            v16h bf = *(const v16h*)Brow;
            acc[t] = __builtin_amdgcn_wmma_f32_16x16x32_f16(
                false, af, false, bf, (short)0, acc[t], false, false);
        }
    }

#pragma unroll
    for (int t = 0; t < 8; ++t) {
#pragma unroll
        for (int v = 0; v < 8; ++v) {
            // f32 C/D layout: lane L holds (M = v + (L/16)*8, N = L%16)
            const int row = rowBase + hi * 8 + v;
            const int col = colChunk + t * 16 + lp;
            float x = acc[t][v];
            if (MODE == 0) {
                ((_Float16*)outp)[(size_t)col * outStride + row] = (_Float16)x;
            } else if (MODE == 1) {
                float y = x > 0.f ? x : (__expf(x) - 1.f);   // per-head ELU
                y = y > 0.f ? y : (__expf(y) - 1.f);         // post-concat ELU
                ((_Float16*)outp)[(size_t)row * outStride + colOffset + col] =
                    (_Float16)y;
            } else {
                ((float*)outp)[(size_t)row * outStride + col] = x;
            }
        }
    }
}

// f32 -> f16 elementwise convert
__global__ __launch_bounds__(256)
void cvt_f16(const float* __restrict__ in, _Float16* __restrict__ out, int n)
{
    int i = blockIdx.x * 256 + threadIdx.x;
    if (i < n) out[i] = (_Float16)in[i];
}

// Transpose-convert one head's weight W[K x D] (f32) -> WT[D x K] (f16)
__global__ __launch_bounds__(256)
void cvt_wT(const float* __restrict__ W, _Float16* __restrict__ WT, int K, int Dd)
{
    int i = blockIdx.x * 256 + threadIdx.x;
    if (i < K * Dd) {
        int k = i % K, d = i / K;
        WT[i] = (_Float16)W[(size_t)k * Dd + d];
    }
}

// f1[n] = Wh[n,:] . a[0:D]   f2[n] = Wh[n,:] . a[D:2D]   (Wh given transposed)
__global__ __launch_bounds__(256)
void fkern(const _Float16* __restrict__ WhT, const float* __restrict__ a,
           int D, int N, float* __restrict__ f1, float* __restrict__ f2)
{
    int n = blockIdx.x * 256 + threadIdx.x;
    float s1 = 0.f, s2 = 0.f;
    for (int d = 0; d < D; ++d) {
        float w = (float)WhT[(size_t)d * N + n];
        s1 += w * a[d];
        s2 += w * a[D + d];
    }
    f1[n] = s1;
    f2[n] = s2;
}

// Per-row online softmax stats over masked leaky-relu logits. One wave per row.
__global__ __launch_bounds__(256)
void stats(const float* __restrict__ adj, const float* __restrict__ f1,
           const float* __restrict__ f2, float* __restrict__ rmax,
           float* __restrict__ rinv, int N)
{
    int row  = blockIdx.x * 8 + (threadIdx.x >> 5);
    int lane = threadIdx.x & 31;
    const float* arow = adj + (size_t)row * N;
    float f1n = f1[row];
    float fm = -3.0e38f, fs = 0.f;
    for (int m = lane; m < N; m += 32) {
        float a = arow[m];
        if (a > 0.f) {
            float e = f1n + f2[m];
            e = e > 0.f ? e : GALPHA * e;
            if (e > fm) { fs = fs * __expf(fm - e) + 1.f; fm = e; }
            else        { fs += __expf(e - fm); }
        }
    }
    for (int off = 16; off; off >>= 1) {
        float om = __shfl_xor(fm, off, 32);
        float os = __shfl_xor(fs, off, 32);
        float nm = fmaxf(fm, om);
        fs = fs * __expf(fm - nm) + os * __expf(om - nm);
        fm = nm;
    }
    if (lane == 0) {
        if (fs > 0.f) { rmax[row] = fm; rinv[row] = 1.f / fs; }
        else { rmax[row] = -3.0e38f; rinv[row] = 1.f / (float)N; } // no neighbors
    }
}

// Materialize softmax attention matrix P[N x N] in f16 (4 cols per thread)
__global__ __launch_bounds__(256)
void pkern(const float* __restrict__ adj, const float* __restrict__ f1,
           const float* __restrict__ f2, const float* __restrict__ rmax,
           const float* __restrict__ rinv, _Float16* __restrict__ P, int N)
{
    int idx = blockIdx.x * 256 + threadIdx.x;
    int perRow = N >> 2;
    int n  = idx / perRow;
    int m0 = (idx - n * perRow) << 2;
    float4 av = reinterpret_cast<const float4*>(adj + (size_t)n * N)[m0 >> 2];
    float mx = rmax[n], inv = rinv[n], f1n = f1[n];
    bool uni = (mx <= -1.0e30f);                  // isolated row -> uniform
    float aa[4] = {av.x, av.y, av.z, av.w};
    v4h pv;
#pragma unroll
    for (int j = 0; j < 4; ++j) {
        float p;
        if (uni) p = inv;
        else if (aa[j] > 0.f) {
            float e = f1n + f2[m0 + j];
            e = e > 0.f ? e : GALPHA * e;
            p = __expf(e - mx) * inv;
        } else p = 0.f;
        pv[j] = (_Float16)p;
    }
    *reinterpret_cast<v4h*>(P + (size_t)n * N + m0) = pv;
}

// mean over 6 heads + ELU + row L2 normalize.  One block per row.
__global__ __launch_bounds__(256)
void finalize(const float* __restrict__ out3, float* __restrict__ out,
              int N, int C)
{
    __shared__ float red[256];
    int n = blockIdx.x;
    float vals[2];
    float ss = 0.f;
#pragma unroll
    for (int j = 0; j < 2; ++j) {
        int c = threadIdx.x + j * 256;
        float s = 0.f;
#pragma unroll
        for (int h = 0; h < 6; ++h)
            s += out3[((size_t)h * N + n) * C + c];
        s *= (1.f / 6.f);
        s = s > 0.f ? s : (__expf(s) - 1.f);
        vals[j] = s;
        ss += s * s;
    }
    red[threadIdx.x] = ss;
    __syncthreads();
    for (int o = 128; o; o >>= 1) {
        if (threadIdx.x < o) red[threadIdx.x] += red[threadIdx.x + o];
        __syncthreads();
    }
    float d = fmaxf(sqrtf(red[0]), 1e-12f);
#pragma unroll
    for (int j = 0; j < 2; ++j) {
        int c = threadIdx.x + j * 256;
        out[(size_t)n * C + c] = vals[j] / d;
    }
}

extern "C" void kernel_launch(void* const* d_in, const int* in_sizes, int n_in,
                              void* d_out, int out_size, void* d_ws, size_t ws_size,
                              hipStream_t stream)
{
    (void)in_sizes; (void)n_in; (void)out_size; (void)ws_size;
    const float* x   = (const float*)d_in[0];   // [N, NFEAT]
    const float* adj = (const float*)d_in[1];   // [N, N]
    const float* W1  = (const float*)d_in[2];   // [4, NFEAT, D]
    const float* a1  = (const float*)d_in[3];   // [4, 2D]
    const float* W2  = (const float*)d_in[4];   // [4, 4D, D]
    const float* a2  = (const float*)d_in[5];   // [4, 2D]
    const float* W3  = (const float*)d_in[6];   // [6, 4D, C]
    const float* a3  = (const float*)d_in[7];   // [6, 2C]
    float* out = (float*)d_out;                 // [N, C]

    const int N = GN, NF = GNFEAT, D = GD, C = GC;
    const int HID = 4 * D; // 1024, concat width

    // ---- workspace layout ----
    char* ws = (char*)d_ws;
    size_t off = 0;
    auto take = [&](size_t bytes) -> char* {
        char* p = ws + off;
        off += (bytes + 255) & ~(size_t)255;
        return p;
    };
    _Float16* x16  = (_Float16*)take((size_t)N * NF * 2);      // 16 MB
    _Float16* h1   = (_Float16*)take((size_t)N * HID * 2);     //  8 MB
    _Float16* h2   = (_Float16*)take((size_t)N * HID * 2);     //  8 MB
    _Float16* w1t  = (_Float16*)take((size_t)4 * D * NF * 2);  //  4 MB
    _Float16* w2t  = (_Float16*)take((size_t)4 * D * HID * 2); //  2 MB
    _Float16* w3t  = (_Float16*)take((size_t)6 * C * HID * 2); //  6 MB
    _Float16* whT  = (_Float16*)take((size_t)C * N * 2);       //  4 MB (max head)
    _Float16* P    = (_Float16*)take((size_t)N * N * 2);       // 32 MB
    float*    f1v  = (float*)take((size_t)N * 4);
    float*    f2v  = (float*)take((size_t)N * 4);
    float*    rmax = (float*)take((size_t)N * 4);
    float*    rinv = (float*)take((size_t)N * 4);
    float*    out3 = (float*)take((size_t)6 * N * C * 4);      // 48 MB

    // ---- prep: convert activations & weights (weights transposed) ----
    cvt_f16<<<(N * NF) / 256, 256, 0, stream>>>(x, x16, N * NF);
    for (int h = 0; h < 4; ++h)
        cvt_wT<<<(NF * D) / 256, 256, 0, stream>>>(W1 + (size_t)h * NF * D,
                                                   w1t + (size_t)h * D * NF, NF, D);
    for (int h = 0; h < 4; ++h)
        cvt_wT<<<(HID * D) / 256, 256, 0, stream>>>(W2 + (size_t)h * HID * D,
                                                    w2t + (size_t)h * D * HID, HID, D);
    for (int h = 0; h < 6; ++h)
        cvt_wT<<<(HID * C) / 256, 256, 0, stream>>>(W3 + (size_t)h * HID * C,
                                                    w3t + (size_t)h * C * HID, HID, C);

    const dim3 blk(256);
    // ---- layer 1: input x16 [N, NF], 4 heads, concat -> h1 [N, HID] ----
    for (int h = 0; h < 4; ++h) {
        gemm_wmma<0><<<dim3(N / 128, D / 128), blk, 0, stream>>>(
            x16, w1t + (size_t)h * D * NF, whT, N, NF, N, 0);
        fkern<<<N / 256, blk, 0, stream>>>(whT, a1 + (size_t)h * 2 * D, D, N, f1v, f2v);
        stats<<<N / 8, blk, 0, stream>>>(adj, f1v, f2v, rmax, rinv, N);
        pkern<<<(N * (N / 4)) / 256, blk, 0, stream>>>(adj, f1v, f2v, rmax, rinv, P, N);
        gemm_wmma<1><<<dim3(N / 128, D / 128), blk, 0, stream>>>(
            P, whT, h1, N, N, HID, h * D);
    }
    // ---- layer 2: input h1 [N, HID], 4 heads, concat -> h2 [N, HID] ----
    for (int h = 0; h < 4; ++h) {
        gemm_wmma<0><<<dim3(N / 128, D / 128), blk, 0, stream>>>(
            h1, w2t + (size_t)h * D * HID, whT, N, HID, N, 0);
        fkern<<<N / 256, blk, 0, stream>>>(whT, a2 + (size_t)h * 2 * D, D, N, f1v, f2v);
        stats<<<N / 8, blk, 0, stream>>>(adj, f1v, f2v, rmax, rinv, N);
        pkern<<<(N * (N / 4)) / 256, blk, 0, stream>>>(adj, f1v, f2v, rmax, rinv, P, N);
        gemm_wmma<1><<<dim3(N / 128, D / 128), blk, 0, stream>>>(
            P, whT, h2, N, N, HID, h * D);
    }
    // ---- layer 3: input h2 [N, HID], 6 heads, mean -> out3 [6, N, C] ----
    for (int h = 0; h < 6; ++h) {
        gemm_wmma<0><<<dim3(N / 128, C / 128), blk, 0, stream>>>(
            h2, w3t + (size_t)h * C * HID, whT, N, HID, N, 0);
        fkern<<<N / 256, blk, 0, stream>>>(whT, a3 + (size_t)h * 2 * C, C, N, f1v, f2v);
        stats<<<N / 8, blk, 0, stream>>>(adj, f1v, f2v, rmax, rinv, N);
        pkern<<<(N * (N / 4)) / 256, blk, 0, stream>>>(adj, f1v, f2v, rmax, rinv, P, N);
        gemm_wmma<2><<<dim3(N / 128, C / 128), blk, 0, stream>>>(
            P, whT, out3 + (size_t)h * N * C, N, N, C, 0);
    }
    // ---- mean + ELU + L2 normalize ----
    finalize<<<N, blk, 0, stream>>>(out3, out, N, C);
}
